// Ranker_25022479466555
// MI455X (gfx1250) — matrix operand order repaired and checked by reading, not compiled, not run on hardware
//
#include <hip/hip_runtime.h>
#include <hip/hip_bf16.h>
#include <math.h>

// Problem constants (fixed by the reference harness)
#define BB  2
#define TT  8192
#define EE  256
#define CS  128
#define KK  8
#define NN  (TT / CS)   // 64
#define LL  (KK * CS)   // 1024
#define EPSF 1e-6f

typedef __attribute__((ext_vector_type(16))) __bf16 v16bf;
typedef __attribute__((ext_vector_type(8)))  __bf16 v8bf;
typedef __attribute__((ext_vector_type(8)))  float  v8f;
typedef __attribute__((ext_vector_type(4)))  unsigned int v4u;
typedef __attribute__((ext_vector_type(8)))  unsigned int v8u;

// ---------------------------------------------------------------------------
// Fragment loaders following CDNA5 ISA 7.12.2 (wave32):
//   A (16x32 bf16, MxK): lane l holds row M=l&15.
//     lanes 0-15:  K = {k0+0..7, k0+16..23}; lanes 16-31: K = {k0+8..15, k0+24..31}
//   B (32x16 bf16, KxN): lane l holds column N=l&15.
//     lanes 0-15: K = {k0+0..7, k0+16..23}; lanes 16-31: K = {k0+8..15, k0+24..31}
// ---------------------------------------------------------------------------
static __device__ __forceinline__ v16bf load_a_frag(const __bf16* base, int ld,
                                                    int m0, int k0, int lane) {
  const int hi = (lane >> 4) & 1;
  const __bf16* row = base + (size_t)(m0 + (lane & 15)) * ld + k0 + hi * 8;
  v8bf lo = *(const v8bf*)(row);
  v8bf hs = *(const v8bf*)(row + 16);
  return __builtin_shufflevector(lo, hs, 0,1,2,3,4,5,6,7,8,9,10,11,12,13,14,15);
}

static __device__ __forceinline__ v16bf load_b_frag(const __bf16* base, int ld,
                                                    int k0, int n0, int lane) {
  const int hi = (lane >> 4) & 1;
  const __bf16* p = base + (size_t)(n0 + (lane & 15)) * ld + k0 + hi * 8;
  v8bf lo = *(const v8bf*)(p);
  v8bf hs = *(const v8bf*)(p + 16);
  return __builtin_shufflevector(lo, hs, 0,1,2,3,4,5,6,7,8,9,10,11,12,13,14,15);
}

// ---------------------------------------------------------------------------
// Kernel 1: per-row L2 norm -> cn (bf16, [B*N*CS, E]) and transposed raw
// chunks XT (bf16, [B*N, E, CS]) for the K-contiguous projection B operand.
// ---------------------------------------------------------------------------
__global__ __launch_bounds__(256) void prep_kernel(const float* __restrict__ x,
                                                   __bf16* __restrict__ cn,
                                                   __bf16* __restrict__ xt) {
  const int row = blockIdx.x;           // b*N*CS + n*CS + r
  const int e   = threadIdx.x;          // 0..255
  const float v = x[(size_t)row * EE + e];

  __shared__ float red[256];
  red[e] = v * v;
  __syncthreads();
  for (int s = 128; s > 0; s >>= 1) {
    if (e < s) red[e] += red[e + s];
    __syncthreads();
  }
  const float inv = 1.0f / (sqrtf(red[0]) + EPSF);

  cn[(size_t)row * EE + e] = (__bf16)(v * inv);

  const int r  = row & (CS - 1);
  const int bn = row / CS;
  xt[((size_t)bn * EE + e) * CS + r] = (__bf16)v;
}

// Kernel 2: down_proj fp32 -> bf16
__global__ void dpconv_kernel(const float* __restrict__ dp, __bf16* __restrict__ dpb) {
  int i = blockIdx.x * blockDim.x + threadIdx.x;
  if (i < CS * LL) dpb[i] = (__bf16)dp[i];
}

// ---------------------------------------------------------------------------
// Kernel 3: causal pair scores.
// Block = one (b,i,j) with j<i. cn_j (128x256 bf16 = 64 KB) staged into LDS by
// the Tensor Data Mover (wave 0 issues tensor_load_to_lds, waits TENSORcnt,
// barrier publishes to the other 7 waves). Wave w owns rows [16w,16w+16) of
// chunk i; per-lane running max over the 8 column tiles, single cross-lane
// max reduction at the end (C/D layout: lanes 0-15 <-> M=r, 16-31 <-> M=8+r).
// ---------------------------------------------------------------------------
__global__ __launch_bounds__(256) void scores_kernel(const __bf16* __restrict__ cn,
                                                     float* __restrict__ scores) {
  const int j = blockIdx.x, i = blockIdx.y, b = blockIdx.z;
  if (j >= i) return;  // uniform exit; causal mask

  __shared__ __align__(16) __bf16 smem[CS * EE];   // cn_j: 64 KB
  __shared__ float sScore;

  const __bf16* cni = cn + ((size_t)(b * NN + i)) * CS * EE;
  const __bf16* cnj = cn + ((size_t)(b * NN + j)) * CS * EE;
  __builtin_prefetch(cni, 0, 1);

  if ((threadIdx.x >> 5) == 0) {   // wave 0 issues the TDM (EXEC ignored)
    // Tensor DMA descriptor (CDNA5 ISA):
    // 2D tile 256x128 of 2-byte elements, contiguous, global -> LDS.
    const unsigned long long ga = (unsigned long long)(uintptr_t)cnj;
    const unsigned int lds_off = (unsigned int)(uintptr_t)smem; // low 32 = LDS byte addr
    v4u g0;
    g0[0] = 1u;                                            // count=1 (valid D#)
    g0[1] = lds_off;                                       // lds_addr [63:32]
    g0[2] = (unsigned int)ga;                              // global_addr lo
    g0[3] = (unsigned int)((ga >> 32) & 0x01FFFFFFu) | (2u << 30); // addr hi, type=2
    v8u g1;
    g1[0] = (1u << 16);        // workgroup_mask=0, data_size=1 (2 bytes)
    g1[1] = (256u << 16);      // tensor_dim0 = 256 (bits 63:48)
    g1[2] = (128u << 16);      // tensor_dim0 hi=0; tensor_dim1 = 128 (bits 95:80)
    g1[3] = (256u << 16);      // tensor_dim1 hi=0; tile_dim0 = 256 (bits 127:112)
    g1[4] = 128u;              // tile_dim1 = 128 (bits 143:128)
    g1[5] = 256u;              // tensor_dim0_stride = 256 (bits 191:160)
    g1[6] = 0u;
    g1[7] = 0u;
    asm volatile("tensor_load_to_lds %0, %1" :: "s"(g0), "s"(g1) : "memory");
    __builtin_amdgcn_s_wait_tensorcnt(0);
    if (threadIdx.x == 0) sScore = 0.0f;
  }
  __syncthreads();

  const int lane = threadIdx.x & 31;
  const int wave = threadIdx.x >> 5;   // 0..7
  const int m0   = wave * 16;

  v16bf a[8];
#pragma unroll
  for (int k = 0; k < 8; ++k) a[k] = load_a_frag(cni, EE, m0, k * 32, lane);

  float rm[8];
#pragma unroll
  for (int r = 0; r < 8; ++r) rm[r] = -1e30f;

  for (int d = 0; d < 8; ++d) {
    v8f c = {};
#pragma unroll
    for (int k = 0; k < 8; ++k) {
      v16bf bf = load_b_frag(smem, EE, k * 32, d * 16, lane);
      c = __builtin_amdgcn_wmma_f32_16x16x32_bf16(false, a[k], false, bf,
                                                  (short)0, c, false, false);
    }
    // per-lane running max over this lane's column (n = lane&15 of tile d)
#pragma unroll
    for (int r = 0; r < 8; ++r) rm[r] = fmaxf(rm[r], c[r]);
  }
  // one cross-lane reduction: max over the 16 lanes of each half -> row max
#pragma unroll
  for (int r = 0; r < 8; ++r) {
    float v = rm[r];
    v = fmaxf(v, __shfl_xor(v, 1, 32));
    v = fmaxf(v, __shfl_xor(v, 2, 32));
    v = fmaxf(v, __shfl_xor(v, 4, 32));
    v = fmaxf(v, __shfl_xor(v, 8, 32));   // stays within 16-lane half
    rm[r] = v;
  }
  float s = 0.0f;
#pragma unroll
  for (int r = 0; r < 8; ++r) s += rm[r];   // 8 rows of this lane's half
  s += __shfl_xor(s, 16, 32);               // + other half -> 16 rows
  if (lane == 0) atomicAdd(&sScore, s);
  __syncthreads();
  if (threadIdx.x == 0) scores[((size_t)b * NN + i) * NN + j] = sScore;
}

// ---------------------------------------------------------------------------
// Kernel 4: top-(K-1) selection, weights, left-pad arrangement.
// ---------------------------------------------------------------------------
__global__ void topk_kernel(const float* __restrict__ scores,
                            float* __restrict__ selw, int* __restrict__ seli) {
  const int t = blockIdx.x * blockDim.x + threadIdx.x;
  if (t >= BB * NN) return;
  const int b = t / NN, i = t % NN;
  const float* row = scores + ((size_t)b * NN + i) * NN;

  float vals[KK - 1]; int idx[KK - 1];
  unsigned long long used = 0ull;
  const int ns = (i < KK - 1) ? i : (KK - 1);
#pragma unroll
  for (int s = 0; s < KK - 1; ++s) { vals[s] = 0.0f; idx[s] = 0; }
  for (int s = 0; s < ns; ++s) {
    float best = -1e30f; int bj = 0;
    for (int jj = 0; jj < i; ++jj)
      if (!((used >> jj) & 1ull) && row[jj] > best) { best = row[jj]; bj = jj; }
    used |= (1ull << bj);
    vals[s] = best; idx[s] = bj;
  }
  float w[KK - 1];
  if (ns > 0) {
    const float vmin = vals[ns - 1];
#pragma unroll
    for (int s = 0; s < KK - 1; ++s)
      w[s] = (s < ns) ? vals[s] / (vmin + EPSF) : 0.0f;
  } else {
#pragma unroll
    for (int s = 0; s < KK - 1; ++s) w[s] = 0.0f;
  }
  const int shift = (KK - 1 - i) > 0 ? (KK - 1 - i) : 0;
#pragma unroll
  for (int s = 0; s < KK - 1; ++s) {
    int src = s - shift;
    src = src < 0 ? 0 : (src > KK - 2 ? KK - 2 : src);
    const bool keep = (s >= shift);
    selw[(size_t)t * (KK - 1) + s] = keep ? w[src] : 0.0f;
    seli[(size_t)t * (KK - 1) + s] = keep ? idx[src] : 0;
  }
}

// ---------------------------------------------------------------------------
// Kernel 5: out[b,n] = down_proj @ ext + chunk, folded as
//   out = sum_s w_s * (dp[:, s*128:(s+1)*128] @ X[idx_s])
// Block = (b,n); 8 waves. e-tiles processed in groups of 4 (4 accumulators)
// so each slot's 4 A-fragments of down_proj are loaded once per group and
// reused across 4 e-tiles (4x less A vmem). Fused +x epilogue in f32.
// ---------------------------------------------------------------------------
__global__ __launch_bounds__(256) void out_kernel(const float* __restrict__ x,
                                                  const __bf16* __restrict__ dpb,
                                                  const __bf16* __restrict__ xt,
                                                  const float* __restrict__ selw,
                                                  const int* __restrict__ seli,
                                                  float* __restrict__ out) {
  const int n = blockIdx.x, b = blockIdx.y;
  const int bn = b * NN + n;
  const int lane = threadIdx.x & 31;
  const int wave = threadIdx.x >> 5;
  const int m0 = wave * 16;

  float w[KK]; int id[KK];
#pragma unroll
  for (int s = 0; s < KK - 1; ++s) {
    w[s]  = selw[(size_t)bn * (KK - 1) + s];   // same value for all lanes -> uniform
    id[s] = seli[(size_t)bn * (KK - 1) + s];
  }
  w[KK - 1] = 1.0f; id[KK - 1] = n;            // current chunk slot

  for (int g = 0; g < 4; ++g) {                // 4 groups of 4 e-tiles
    v8f acc[4];
#pragma unroll
    for (int u = 0; u < 4; ++u) acc[u] = (v8f){};

#pragma unroll
    for (int s = 0; s < KK; ++s) {
      if (w[s] == 0.0f) continue;              // block-uniform branch
      const __bf16* xtc = xt + ((size_t)(b * NN + id[s])) * EE * CS;
      v16bf a[4];
#pragma unroll
      for (int kk = 0; kk < 4; ++kk)
        a[kk] = load_a_frag(dpb, LL, m0, s * CS + kk * 32, lane);
#pragma unroll
      for (int u = 0; u < 4; ++u) {
        const int e0 = (g * 4 + u) * 16;
        v8f p = {};
#pragma unroll
        for (int kk = 0; kk < 4; ++kk) {
          v16bf bf = load_b_frag(xtc, CS, kk * 32, e0, lane);
          p = __builtin_amdgcn_wmma_f32_16x16x32_bf16(false, a[kk], false, bf,
                                                      (short)0, p, false, false);
        }
#pragma unroll
        for (int r = 0; r < 8; ++r) acc[u][r] += w[s] * p[r];
      }
    }
    const int hi = (lane >> 4) & 1;
#pragma unroll
    for (int u = 0; u < 4; ++u) {
      const int e = (g * 4 + u) * 16 + (lane & 15);
#pragma unroll
      for (int r = 0; r < 8; ++r) {
        const int m = m0 + r + 8 * hi;         // C/D VGPR layout
        const size_t oi = ((size_t)b * TT + (size_t)n * CS + m) * EE + e;
        out[oi] = acc[u][r] + x[oi];
      }
    }
  }
}

// ---------------------------------------------------------------------------
extern "C" void kernel_launch(void* const* d_in, const int* in_sizes, int n_in,
                              void* d_out, int out_size, void* d_ws, size_t ws_size,
                              hipStream_t stream) {
  const float* x  = (const float*)d_in[0];   // [B,T,E] f32
  const float* dp = (const float*)d_in[1];   // [CS,L] f32
  float* out = (float*)d_out;                // [B,T,E] f32

  char* ws = (char*)d_ws;
  size_t off = 0;
  auto take = [&](size_t bytes) -> char* {
    char* p = ws + off;
    off = (off + bytes + 255) & ~(size_t)255;
    return p;
  };
  __bf16* cn   = (__bf16*)take((size_t)BB * NN * CS * EE * sizeof(__bf16)); // 8 MB
  __bf16* xt   = (__bf16*)take((size_t)BB * NN * EE * CS * sizeof(__bf16)); // 8 MB
  __bf16* dpb  = (__bf16*)take((size_t)CS * LL * sizeof(__bf16));           // 256 KB
  float*  sc   = (float*)take((size_t)BB * NN * NN * sizeof(float));        // 32 KB
  float*  selw = (float*)take((size_t)BB * NN * (KK - 1) * sizeof(float));
  int*    seli = (int*)take((size_t)BB * NN * (KK - 1) * sizeof(int));

  prep_kernel<<<BB * NN * CS, 256, 0, stream>>>(x, cn, xt);
  dpconv_kernel<<<(CS * LL + 255) / 256, 256, 0, stream>>>(dp, dpb);
  scores_kernel<<<dim3(NN, NN, BB), 256, 0, stream>>>(cn, sc);
  topk_kernel<<<1, 128, 0, stream>>>(sc, selw, seli);
  out_kernel<<<dim3(NN, BB), 256, 0, stream>>>(x, dpb, xt, selw, seli, out);
}
